// SpatioTemporalBlock_19061064859665
// MI455X (gfx1250) — matrix operand order repaired and checked by reading, not compiled, not run on hardware
//
#include <hip/hip_runtime.h>
#include <hip/hip_bf16.h>

// ---------------- problem constants ----------------
constexpr int cB = 2, cN = 1000, cF = 8, cT = 12, cH = 64;
constexpr int cNH = 4, cDH = 16;
constexpr int cER = 2000, cEC = 24000;
constexpr int cBT = cB * cT;            // 24
constexpr int cNF = cN * cF;            // 8000
constexpr int cM1 = cBT * cN;           // 24000  (river rows)
constexpr int cM2 = cBT * cNF;          // 192000 (causal rows)
constexpr int cR  = cB * cNF;           // 16000  (sequences)
constexpr int cRT = cR * cT;            // 192000 (seq rows)

typedef _Float16 h16_t __attribute__((ext_vector_type(16)));
typedef _Float16 h8_t  __attribute__((ext_vector_type(8)));
typedef float    f8_t  __attribute__((ext_vector_type(8)));

// ---------------- workspace arena (bytes) ----------------
// Wbuf: transposed f16 weights.
constexpr size_t OFF_WBUF   = 0;
constexpr size_t OFF_CATR   = 262144;                       // f16 [M1,128]  (aggr_r | ws)
constexpr size_t OFF_LINR   = OFF_CATR   + (size_t)cM1*128*2; // f16 [M1,64]
constexpr size_t OFF_AGGR_R = OFF_LINR   + (size_t)cM1*64*2;  // f32 [M1,64]
constexpr size_t OFF_WSU    = OFF_AGGR_R + (size_t)cM1*64*4;  // f16 [M1,64]
constexpr size_t OFF_B0     = OFF_WSU    + (size_t)cM1*64*2;  // cat_c f16[M2,128] -> seq f32[RT,64]
constexpr size_t OFF_B1     = OFF_B0     + (size_t)cM2*128*2; // lin_c f16[M2,64]  -> ln_h f16[RT,64]
constexpr size_t OFF_B2     = OFF_B1     + (size_t)cM2*64*2;  // aggr_c f32[M2,64] -> cat_f f16[M2,128] -> o f16[RT,64]
constexpr size_t OFF_B3     = OFF_B2     + (size_t)cM2*64*4;  // fus1 f16[M2,64] -> qkv f16[RT,192] -> ff1 f16[RT,256]

// weight offsets inside Wbuf (in halves), all stored transposed [Nout, K]
constexpr size_t WT_RVLIN = 0;       // [64,64]
constexpr size_t WT_RVUPD = 4096;    // [64,128]
constexpr size_t WT_CSLIN = 12288;   // [64,64]
constexpr size_t WT_CSUPD = 16384;   // [64,128]
constexpr size_t WT_FU1   = 24576;   // [64,128]
constexpr size_t WT_FU2   = 32768;   // [64,64]
constexpr size_t WT_QKV   = 36864;   // [192,64]
constexpr size_t WT_O     = 49152;   // [64,64]
constexpr size_t WT_FF1   = 53248;   // [256,64]
constexpr size_t WT_FF2   = 69632;   // [64,256]

// ---------------- weight transpose+convert:  W[K,Nout] f32 -> Wt[Nout,K] f16 ----------------
__global__ void wconv_kernel(const float* __restrict__ W, _Float16* __restrict__ Wt,
                             int K, int Nout) {
    int i = blockIdx.x * blockDim.x + threadIdx.x;
    if (i >= K * Nout) return;
    int n = i / K, k = i - n * K;
    Wt[i] = (_Float16)W[(size_t)k * Nout + n];
}

// ---------------- prep: x[B,N,F,T,H] -> cat_c[:,64:] (fn f16), cat_r[:,64:] (ws f16) ----------------
__global__ void prep_kernel(const float* __restrict__ x,
                            _Float16* __restrict__ cat_c, _Float16* __restrict__ cat_r) {
    int tid = blockIdx.x * blockDim.x + threadIdx.x;
    if (tid >= cBT * cN * cH) return;
    int h  = tid & 63;
    int n  = (tid >> 6) % cN;
    int bt = tid / (64 * cN);
    int b = bt / cT, t = bt - b * cT;
    float s = 0.f;
#pragma unroll
    for (int f = 0; f < cF; ++f) {
        float v = x[((((size_t)b * cN + n) * cF + f) * cT + t) * cH + h];
        cat_c[((size_t)bt * cNF + n * cF + f) * 128 + 64 + h] = (_Float16)v;
        s += v;
    }
    cat_r[((size_t)bt * cN + n) * 128 + 64 + h] = (_Float16)(s * 0.125f);
}

// ---------------- gated edge scatter-add (message passing, aggr='add') ----------------
__global__ void scatter_kernel(const _Float16* __restrict__ lin, const int* __restrict__ eidx,
                               const float* __restrict__ ew, const float* __restrict__ lw,
                               const float* __restrict__ gatep, float* __restrict__ aggr,
                               int E, int nNodes) {
    int tid = blockIdx.x * blockDim.x + threadIdx.x;
    int total = cBT * E * 16;
    if (tid >= total) return;
    int h4 = tid & 15;
    int e  = (tid >> 4) % E;
    int bt = tid / (16 * E);
    float g  = 1.f / (1.f + __expf(-gatep[0]));
    float lg = 1.f / (1.f + __expf(-lw[e]));
    float mw = g * ew[e] + (1.f - g) * lg;
    int src = eidx[e], dst = eidx[E + e];
    const _Float16* lp = lin + ((size_t)bt * nNodes + src) * 64 + h4 * 4;
    float* ap = aggr + ((size_t)bt * nNodes + dst) * 64 + h4 * 4;
#pragma unroll
    for (int i = 0; i < 4; ++i) atomicAdd(ap + i, mw * (float)lp[i]);
}

// ---------------- f32 aggr -> f16 cat[:,0:64] ----------------
__global__ void aggr2cat_kernel(const float* __restrict__ aggr, _Float16* __restrict__ cat,
                                int total) {
    int tid = blockIdx.x * blockDim.x + threadIdx.x;
    if (tid >= total) return;
    int h = tid & 63, m = tid >> 6;
    cat[(size_t)m * 128 + h] = (_Float16)aggr[tid];
}

// ---------------- broadcast ws_u over F into cat_f[:,64:] ----------------
__global__ void bcast_kernel(const _Float16* __restrict__ wsu, _Float16* __restrict__ cat_f) {
    int tid = blockIdx.x * blockDim.x + threadIdx.x;
    if (tid >= cM2 * 64) return;
    int h = tid & 63, m = tid >> 6;
    int bt = m / cNF, nf = m - bt * cNF, n = nf >> 3;  // F = 8
    cat_f[(size_t)m * 128 + 64 + h] = wsu[((size_t)bt * cN + n) * 64 + h];
}

// ---------------- WMMA GEMM:  act(A[M,K](f16) @ Bt[Nout,K]^T(f16) + bias) ----------------
// ACT: 0=none 1=relu 2=gelu(exact). OUTMODE: 1=f16 store, 2=f32 +=, 3=f32 store w/ fused->seq remap
template <int ACT, int OUTMODE>
__global__ __launch_bounds__(256) void gemm_wmma_kernel(
    const _Float16* __restrict__ A, int lda,
    const _Float16* __restrict__ Bt,
    const float* __restrict__ bias,
    void* __restrict__ outp, int ldo,
    int M, int K, int Nout) {
    int wave = threadIdx.x >> 5;
    int lane = threadIdx.x & 31;
    int tilesX = Nout >> 4;
    int tile = blockIdx.x * 8 + wave;
    if (tile >= (M >> 4) * tilesX) return;
    int m0 = (tile / tilesX) << 4;
    int n0 = (tile - (tile / tilesX) * tilesX) << 4;
    int lm = lane & 15;
    int hi = lane >> 4;
    const _Float16* arow = A + (size_t)(m0 + lm) * lda;
    const _Float16* brow = Bt + (size_t)(n0 + lm) * K;
    f8_t acc = {};
    for (int k0 = 0; k0 < K; k0 += 32) {
        // A frag (16x32 f16): halves 0..7 -> K = k0+8*hi+0..7 ; 8..15 -> K = k0+16+8*hi+0..7
        h8_t a0 = *(const h8_t*)(arow + k0 + 8 * hi);
        h8_t a1 = *(const h8_t*)(arow + k0 + 16 + 8 * hi);
        // B frag (32x16 f16): halves 0..7 -> K = k0+8*hi+0..7 ; 8..15 -> K = k0+16+8*hi+0..7, column n = n0+lm
        h8_t b0 = *(const h8_t*)(brow + k0 + 8 * hi);
        h8_t b1 = *(const h8_t*)(brow + k0 + 16 + 8 * hi);
        h16_t a, b;
#pragma unroll
        for (int i = 0; i < 8; ++i) {
            a[i] = a0[i]; a[8 + i] = a1[i];
            b[i] = b0[i]; b[8 + i] = b1[i];
        }
        acc = __builtin_amdgcn_wmma_f32_16x16x32_f16(false, a, false, b, (short)0, acc,
                                                     false, false);
    }
    int ncol = n0 + lm;
    float bv = bias ? bias[ncol] : 0.f;
#pragma unroll
    for (int v = 0; v < 8; ++v) {
        int m = m0 + v + 8 * hi;  // C/D layout: VGPR v, lanes16-31 -> M = v + 8
        float val = acc[v] + bv;
        if (ACT == 1) val = fmaxf(val, 0.f);
        if (ACT == 2) val = 0.5f * val * (1.f + erff(val * 0.70710678118f));
        if (OUTMODE == 1) {
            ((_Float16*)outp)[(size_t)m * ldo + ncol] = (_Float16)val;
        } else if (OUTMODE == 2) {
            float* o = (float*)outp;
            o[(size_t)m * ldo + ncol] += val;
        } else {  // OUTMODE == 3: fused[bt*NF+nf] -> seq[(b*NF+nf)*T + t]
            int bt = m / cNF, nf = m - bt * cNF;
            int b = bt / cT, t = bt - b * cT;
            ((float*)outp)[((size_t)(b * cNF + nf) * cT + t) * 64 + ncol] = val;
        }
    }
}

// ---------------- layernorm (row of 64, f32 in -> f16 out) ----------------
__global__ void ln_kernel(const float* __restrict__ in, const float* __restrict__ g,
                          const float* __restrict__ bb, _Float16* __restrict__ out, int rows) {
    int r = blockIdx.x * blockDim.x + threadIdx.x;
    if (r >= rows) return;
    const float* p = in + (size_t)r * 64;
    float m = 0.f;
#pragma unroll
    for (int i = 0; i < 64; ++i) m += p[i];
    m *= (1.f / 64.f);
    float v = 0.f;
#pragma unroll
    for (int i = 0; i < 64; ++i) { float d = p[i] - m; v += d * d; }
    v *= (1.f / 64.f);
    float inv = rsqrtf(v + 1e-5f);
    _Float16* o = out + (size_t)r * 64;
#pragma unroll
    for (int i = 0; i < 64; ++i) o[i] = (_Float16)((p[i] - m) * inv * g[i] + bb[i]);
}

// ---------------- attention (T=12, 4 heads, DH=16): qkv f16 [RT,192] -> o f16 [RT,64] ----------------
__global__ void attn_kernel(const _Float16* __restrict__ qkv, _Float16* __restrict__ o) {
    int tid = blockIdx.x * blockDim.x + threadIdx.x;
    if (tid >= cR * cT * cNH) return;
    int head = tid & 3;
    int tq   = (tid >> 2) % cT;
    int s    = tid / (cNH * cT);
    const _Float16* base = qkv + (size_t)s * cT * 192;
    const _Float16* q = base + tq * 192 + head * 16;
    float qv[16];
#pragma unroll
    for (int d = 0; d < 16; ++d) qv[d] = (float)q[d];
    float sc[cT];
    float mx = -3.0e30f;
    for (int tk = 0; tk < cT; ++tk) {
        const _Float16* k = base + tk * 192 + 64 + head * 16;
        float a = 0.f;
#pragma unroll
        for (int d = 0; d < 16; ++d) a += qv[d] * (float)k[d];
        a *= 0.25f;  // 1/sqrt(16)
        sc[tk] = a;
        mx = fmaxf(mx, a);
    }
    float sum = 0.f;
    for (int tk = 0; tk < cT; ++tk) { sc[tk] = __expf(sc[tk] - mx); sum += sc[tk]; }
    float inv = 1.f / sum;
    float ov[16] = {};
    for (int tk = 0; tk < cT; ++tk) {
        const _Float16* v = base + tk * 192 + 128 + head * 16;
        float w = sc[tk] * inv;
#pragma unroll
        for (int d = 0; d < 16; ++d) ov[d] += w * (float)v[d];
    }
    _Float16* op = o + ((size_t)s * cT + tq) * 64 + head * 16;
#pragma unroll
    for (int d = 0; d < 16; ++d) op[d] = (_Float16)ov[d];
}

// ---------------- mean over time: seq f32 [R,T,64] -> out f32 [B,N,F,64] ----------------
__global__ void mean_kernel(const float* __restrict__ seq, float* __restrict__ out) {
    int tid = blockIdx.x * blockDim.x + threadIdx.x;
    if (tid >= cR * 64) return;
    int h = tid & 63, s = tid >> 6;
    float a = 0.f;
#pragma unroll
    for (int t = 0; t < cT; ++t) a += seq[((size_t)s * cT + t) * 64 + h];
    out[tid] = a * (1.f / 12.f);
}

// ---------------- host launch ----------------
static inline int cdiv(int a, int b) { return (a + b - 1) / b; }

extern "C" void kernel_launch(void* const* d_in, const int* in_sizes, int n_in,
                              void* d_out, int out_size, void* d_ws, size_t ws_size,
                              hipStream_t stream) {
    (void)in_sizes; (void)n_in; (void)out_size; (void)ws_size;
    const float* x       = (const float*)d_in[0];
    const int*   ridx    = (const int*)d_in[1];
    const float* rew     = (const float*)d_in[2];
    const int*   cidx    = (const int*)d_in[3];
    const float* cew     = (const float*)d_in[4];
    const float* rv_Wlin = (const float*)d_in[5];
    const float* rv_blin = (const float*)d_in[6];
    const float* rv_Wupd = (const float*)d_in[7];
    const float* rv_bupd = (const float*)d_in[8];
    const float* rv_gate = (const float*)d_in[9];
    const float* rv_lw   = (const float*)d_in[10];
    const float* cs_Wlin = (const float*)d_in[11];
    const float* cs_blin = (const float*)d_in[12];
    const float* cs_Wupd = (const float*)d_in[13];
    const float* cs_bupd = (const float*)d_in[14];
    const float* cs_gate = (const float*)d_in[15];
    const float* cs_lw   = (const float*)d_in[16];
    const float* fu_W1   = (const float*)d_in[17];
    const float* fu_b1   = (const float*)d_in[18];
    const float* fu_W2   = (const float*)d_in[19];
    const float* fu_b2   = (const float*)d_in[20];
    const float* ln1_g   = (const float*)d_in[21];
    const float* ln1_b   = (const float*)d_in[22];
    const float* ln2_g   = (const float*)d_in[23];
    const float* ln2_b   = (const float*)d_in[24];
    const float* W_qkv   = (const float*)d_in[25];
    const float* b_qkv   = (const float*)d_in[26];
    const float* W_o     = (const float*)d_in[27];
    const float* b_o     = (const float*)d_in[28];
    const float* ff_W1   = (const float*)d_in[29];
    const float* ff_b1   = (const float*)d_in[30];
    const float* ff_W2   = (const float*)d_in[31];
    const float* ff_b2   = (const float*)d_in[32];

    char* ws = (char*)d_ws;
    _Float16* wbuf   = (_Float16*)(ws + OFF_WBUF);
    _Float16* cat_r  = (_Float16*)(ws + OFF_CATR);
    _Float16* lin_r  = (_Float16*)(ws + OFF_LINR);
    float*    aggr_r = (float*)   (ws + OFF_AGGR_R);
    _Float16* wsu    = (_Float16*)(ws + OFF_WSU);
    _Float16* cat_c  = (_Float16*)(ws + OFF_B0);
    float*    seq    = (float*)   (ws + OFF_B0);   // aliases cat_c (dead after upd_c)
    _Float16* lin_c  = (_Float16*)(ws + OFF_B1);
    _Float16* ln_h   = (_Float16*)(ws + OFF_B1);   // aliases lin_c
    float*    aggr_c = (float*)   (ws + OFF_B2);
    _Float16* cat_f  = (_Float16*)(ws + OFF_B2);   // aliases aggr_c (after convert)
    _Float16* o_h    = (_Float16*)(ws + OFF_B2);   // aliases cat_f (after fusion1)
    _Float16* fus1   = (_Float16*)(ws + OFF_B3);
    _Float16* qkv_h  = (_Float16*)(ws + OFF_B3);   // aliases fus1
    _Float16* ff1_h  = (_Float16*)(ws + OFF_B3);   // aliases qkv

    const int TB = 256;
    // -- P0: transpose+convert all weights to f16 [Nout,K]
    struct WSpec { const float* W; size_t off; int K, Nout; };
    const WSpec wl[10] = {
        {rv_Wlin, WT_RVLIN, 64, 64},   {rv_Wupd, WT_RVUPD, 128, 64},
        {cs_Wlin, WT_CSLIN, 64, 64},   {cs_Wupd, WT_CSUPD, 128, 64},
        {fu_W1,   WT_FU1,   128, 64},  {fu_W2,   WT_FU2,   64, 64},
        {W_qkv,   WT_QKV,   64, 192},  {W_o,     WT_O,     64, 64},
        {ff_W1,   WT_FF1,   64, 256},  {ff_W2,   WT_FF2,   256, 64}};
    for (int i = 0; i < 10; ++i)
        wconv_kernel<<<cdiv(wl[i].K * wl[i].Nout, TB), TB, 0, stream>>>(
            wl[i].W, wbuf + wl[i].off, wl[i].K, wl[i].Nout);

    // -- P1: fn -> cat_c[:,64:], ws -> cat_r[:,64:]
    prep_kernel<<<cdiv(cBT * cN * cH, TB), TB, 0, stream>>>(x, cat_c, cat_r);

    auto gemmGrid = [](int M, int Nout) { return cdiv((M >> 4) * (Nout >> 4), 8); };

    // -- river conv
    gemm_wmma_kernel<0, 1><<<gemmGrid(cM1, 64), TB, 0, stream>>>(
        cat_r + 64, 128, wbuf + WT_RVLIN, rv_blin, lin_r, 64, cM1, 64, 64);
    hipMemsetAsync(aggr_r, 0, (size_t)cM1 * 64 * 4, stream);
    scatter_kernel<<<cdiv(cBT * cER * 16, TB), TB, 0, stream>>>(
        lin_r, ridx, rew, rv_lw, rv_gate, aggr_r, cER, cN);
    aggr2cat_kernel<<<cdiv(cM1 * 64, TB), TB, 0, stream>>>(aggr_r, cat_r, cM1 * 64);
    gemm_wmma_kernel<1, 1><<<gemmGrid(cM1, 64), TB, 0, stream>>>(
        cat_r, 128, wbuf + WT_RVUPD, rv_bupd, wsu, 64, cM1, 128, 64);

    // -- causal conv
    gemm_wmma_kernel<0, 1><<<gemmGrid(cM2, 64), TB, 0, stream>>>(
        cat_c + 64, 128, wbuf + WT_CSLIN, cs_blin, lin_c, 64, cM2, 64, 64);
    hipMemsetAsync(aggr_c, 0, (size_t)cM2 * 64 * 4, stream);
    scatter_kernel<<<cdiv(cBT * cEC * 16, TB), TB, 0, stream>>>(
        lin_c, cidx, cew, cs_lw, cs_gate, aggr_c, cEC, cNF);
    aggr2cat_kernel<<<cdiv(cM2 * 64, TB), TB, 0, stream>>>(aggr_c, cat_c, cM2 * 64);
    // upd_c: relu([aggr_c, fn] @ cs_Wupd + b) -> cat_f[:,0:64]  (cat_f aliases aggr_c; GEMM
    // reads cat_c, writes cat_f — but aggr_c was already folded into cat_c, safe)
    gemm_wmma_kernel<1, 1><<<gemmGrid(cM2, 64), TB, 0, stream>>>(
        cat_c, 128, wbuf + WT_CSUPD, cs_bupd, cat_f, 128, cM2, 128, 64);
    bcast_kernel<<<cdiv(cM2 * 64, TB), TB, 0, stream>>>(wsu, cat_f);

    // -- fusion
    gemm_wmma_kernel<1, 1><<<gemmGrid(cM2, 64), TB, 0, stream>>>(
        cat_f, 128, wbuf + WT_FU1, fu_b1, fus1, 64, cM2, 128, 64);
    gemm_wmma_kernel<0, 3><<<gemmGrid(cM2, 64), TB, 0, stream>>>(
        fus1, 64, wbuf + WT_FU2, fu_b2, seq, 64, cM2, 64, 64);

    // -- transformer: pre-norm attention
    ln_kernel<<<cdiv(cRT, TB), TB, 0, stream>>>(seq, ln1_g, ln1_b, ln_h, cRT);
    gemm_wmma_kernel<0, 1><<<gemmGrid(cRT, 192), TB, 0, stream>>>(
        ln_h, 64, wbuf + WT_QKV, b_qkv, qkv_h, 192, cRT, 64, 192);
    attn_kernel<<<cdiv(cR * cT * cNH, TB), TB, 0, stream>>>(qkv_h, o_h);
    gemm_wmma_kernel<0, 2><<<gemmGrid(cRT, 64), TB, 0, stream>>>(
        o_h, 64, wbuf + WT_O, b_o, seq, 64, cRT, 64, 64);  // seq += o @ W_o + b_o

    // -- transformer: pre-norm FFN (exact gelu)
    ln_kernel<<<cdiv(cRT, TB), TB, 0, stream>>>(seq, ln2_g, ln2_b, ln_h, cRT);
    gemm_wmma_kernel<2, 1><<<gemmGrid(cRT, 256), TB, 0, stream>>>(
        ln_h, 64, wbuf + WT_FF1, ff_b1, ff1_h, 256, cRT, 64, 256);
    gemm_wmma_kernel<0, 2><<<gemmGrid(cRT, 64), TB, 0, stream>>>(
        ff1_h, 256, wbuf + WT_FF2, ff_b2, seq, 64, cRT, 256, 64);  // seq += ffn

    // -- mean over time
    mean_kernel<<<cdiv(cR * 64, TB), TB, 0, stream>>>(seq, (float*)d_out);
}
